// HierMatcher_60842506715643
// MI455X (gfx1250) — matrix operand order repaired and checked by reading, not compiled, not run on hardware
//
#include <hip/hip_runtime.h>
#include <hip/hip_bf16.h>

typedef __attribute__((ext_vector_type(16))) _Float16 v16h;
typedef __attribute__((ext_vector_type(8)))  _Float16 v8h;
typedef __attribute__((ext_vector_type(8)))  float    v8f;

#define H_   256
#define E_   768
#define H2_  512
#define G3_  768   // 3*H
#define B_   16
#define LF_  32
#define T_   64    // tokens per side (2 fields * 32)
#define DE_  2048

// Dynamic LDS for the fused token-matching kernel:
// 64x512 f16 cm tile + 64 scores + 64 softmax weights
#define TM_SMEM_BYTES (64 * 512 * 2 + 2 * 64 * 4)

// ---------------------------------------------------------------------------
// WMMA helpers (CDNA5 v_wmma_f32_16x16x32_f16, wave32)
// A-frag (16x32, MxK): lane m = lane&15; K base = 16*(lane>=16);
//   elems 0..15 -> K = base..base+15 (16 contiguous)
// B-frag (32x16, KxN): lane n = lane&15; K = 16*(lane>=16) + e (16 contiguous)
// D-frag (16x16 f32):  elem r -> M = r + 8*(lane>=16), N = lane&15
// ---------------------------------------------------------------------------
__device__ __forceinline__ v8f wmma_f16(v16h a, v16h b, v8f c) {
  return __builtin_amdgcn_wmma_f32_16x16x32_f16(false, a, false, b, (short)0, c,
                                                false, false);
}

__device__ __forceinline__ v16h load_a_frag(const _Float16* A, int lda, int m0,
                                            int k0, int lane) {
  int m  = m0 + (lane & 15);
  int kb = k0 + ((lane >> 4) << 4);
  v8h lo = *(const v8h*)(A + (long)m * lda + kb);
  v8h hi = *(const v8h*)(A + (long)m * lda + kb + 8);
  return __builtin_shufflevector(lo, hi, 0, 1, 2, 3, 4, 5, 6, 7,
                                 8, 9, 10, 11, 12, 13, 14, 15);
}

__device__ __forceinline__ v16h load_b_frag(const _Float16* W, int ldw, int n0,
                                            int k0, int lane) {
  int n  = n0 + (lane & 15);
  int kb = k0 + ((lane & 16) ? 16 : 0);
  return *(const v16h*)(W + (long)n * ldw + kb);
}

__device__ __forceinline__ float sig_(float x) { return 1.0f / (1.0f + expf(-x)); }

// ---------------------------------------------------------------------------
// f32 -> f16 conversion / zero-init
// ---------------------------------------------------------------------------
__global__ void k_f2h(const float* __restrict__ s, _Float16* __restrict__ d, long n) {
  long i = (long)blockIdx.x * blockDim.x + threadIdx.x;
  long st = (long)gridDim.x * blockDim.x;
  for (; i < n; i += st) d[i] = (_Float16)s[i];
}

__global__ void k_zero_h(float* __restrict__ a, _Float16* __restrict__ b, long n) {
  long i = (long)blockIdx.x * blockDim.x + threadIdx.x;
  long st = (long)gridDim.x * blockDim.x;
  for (; i < n; i += st) { a[i] = 0.0f; b[i] = (_Float16)0.0f; }
}

// ---------------------------------------------------------------------------
// Token masks: tok nonzero iff any of E elements != 0.  grid = 4*16*32 blocks.
// ---------------------------------------------------------------------------
__global__ void k_masks(const float* __restrict__ l0, const float* __restrict__ l1,
                        const float* __restrict__ r0, const float* __restrict__ r1,
                        int* __restrict__ maskL, int* __restrict__ maskR) {
  int id = blockIdx.x;
  int t = id & 31; id >>= 5;
  int b = id & 15; id >>= 4;
  int sf = id;  // 0:l0 1:l1 2:r0 3:r1
  const float* src = (sf == 0) ? l0 : (sf == 1) ? l1 : (sf == 2) ? r0 : r1;
  __shared__ int flag;
  if (threadIdx.x == 0) flag = 0;
  __syncthreads();
  const float* row = src + ((long)b * LF_ + t) * E_;
  int any = 0;
  for (int k = threadIdx.x; k < E_; k += blockDim.x) any |= (row[k] != 0.0f);
  if (any) atomicOr(&flag, 1);
  __syncthreads();
  if (threadIdx.x == 0) {
    int* m = (sf < 2) ? maskL : maskR;
    m[b * T_ + (sf & 1) * LF_ + t] = flag;
  }
}

// ---------------------------------------------------------------------------
// Batched GEMM + bias:  C[fd] = A[f] @ W[d]^T + bias[d]
// A: M x K f16 row-major, W: N x K f16 (weights as stored (out,in)), C: M x N f32
// Each wave: 64x16 output (4 M-tiles sharing every B-fragment -> 4x less
// weight traffic per WMMA).  grid.y = 8 (f*2+d).
// ---------------------------------------------------------------------------
__global__ void k_inproj(const _Float16* __restrict__ A0, long aStrideF,
                         const _Float16* __restrict__ W0, long wStrideD,
                         const float* __restrict__ b0, long bStrideD,
                         float* __restrict__ C0, long cStrideFD,
                         int M, int N, int K) {
  const int lane = threadIdx.x & 31;
  const int wave = threadIdx.x >> 5;
  const int fd = blockIdx.y;
  const _Float16* A = A0 + (long)(fd >> 1) * aStrideF;
  const _Float16* W = W0 + (long)(fd & 1) * wStrideD;
  const float* bias = b0 + (long)(fd & 1) * bStrideD;
  float* C = C0 + (long)fd * cStrideFD;
  const int nT = N >> 4;
  int job = blockIdx.x * 8 + wave;
  if (job >= (M >> 6) * nT) return;  // wave-uniform
  int m0 = (job / nT) << 6, n0 = (job % nT) << 4;
  v8f acc[4] = {{}, {}, {}, {}};
  for (int k0 = 0; k0 < K; k0 += 32) {
    v16h bf = load_b_frag(W, K, n0, k0, lane);
#pragma unroll
    for (int mt = 0; mt < 4; ++mt)
      acc[mt] = wmma_f16(load_a_frag(A, K, m0 + mt * 16, k0, lane), bf, acc[mt]);
  }
  int n = n0 + (lane & 15);
  float bv = bias[n];
  int mb = (lane >> 4) << 3;
#pragma unroll
  for (int mt = 0; mt < 4; ++mt)
#pragma unroll
    for (int r = 0; r < 8; ++r)
      C[(long)(m0 + mt * 16 + mb + r) * N + n] = acc[mt][r] + bv;
}

// ---------------------------------------------------------------------------
// One GRU time step for all 8 (field,dir) instances of the current layer.
// gh = h @ Whh^T (+bhh); gates; write h' (f32+f16) and sequence output.
// grid = 32 blocks (inst 0..7 x chunk 0..3), 128 threads (4 waves).
// ---------------------------------------------------------------------------
__global__ void k_gru_step(const float* __restrict__ xi0,       // (8,512,768)
                           const _Float16* __restrict__ Whh0, long wStrideD,  // K=256
                           const float* __restrict__ bhh0, long bhhStrideD,
                           const float* __restrict__ h32p, float* __restrict__ h32n,
                           const _Float16* __restrict__ h16p, _Float16* __restrict__ h16n,
                           float* __restrict__ hout32,          // (4,16,32,512)
                           _Float16* __restrict__ hout16, int t) {
  __shared__ float gh[3][16][64];
  const int lane = threadIdx.x & 31;
  const int wave = threadIdx.x >> 5;  // 0..3
  const int inst = blockIdx.x >> 2;   // 0..7
  const int chunk = blockIdx.x & 3;
  const int f = inst >> 1, d = inst & 1;
  const int j0 = chunk * 64;
  const int tq = d ? (LF_ - 1 - t) : t;
  const float* xi = xi0 + (long)inst * 512 * G3_;
  const _Float16* W = Whh0 + (long)d * wStrideD;
  const float* bhh = bhh0 + (long)d * bhhStrideD;
  const _Float16* hp16 = h16p + (long)inst * B_ * H_;

#pragma unroll
  for (int gi = 0; gi < 3; ++gi) {
    int n0 = gi * H_ + j0 + wave * 16;
    v8f acc = {};
    for (int k0 = 0; k0 < H_; k0 += 32)
      acc = wmma_f16(load_a_frag(hp16, H_, 0, k0, lane),
                     load_b_frag(W, H_, n0, k0, lane), acc);
    int jl = wave * 16 + (lane & 15);
    int mb = (lane >> 4) << 3;
#pragma unroll
    for (int r = 0; r < 8; ++r) gh[gi][mb + r][jl] = acc[r];
  }
  __syncthreads();

  for (int idx = threadIdx.x; idx < 16 * 64; idx += blockDim.x) {
    int b = idx >> 6, jl = idx & 63;
    int j = j0 + jl;
    const float* xir = xi + ((long)b * LF_ + tq) * G3_;
    float r = sig_(xir[j] + gh[0][b][jl] + bhh[j]);
    float z = sig_(xir[H_ + j] + gh[1][b][jl] + bhh[H_ + j]);
    float n = tanhf(xir[2 * H_ + j] + r * (gh[2][b][jl] + bhh[2 * H_ + j]));
    float hp = h32p[((long)inst * B_ + b) * H_ + j];
    float hn = (1.0f - z) * n + z * hp;
    h32n[((long)inst * B_ + b) * H_ + j] = hn;
    h16n[((long)inst * B_ + b) * H_ + j] = (_Float16)hn;
    long o = (((long)(f * B_ + b)) * LF_ + tq) * H2_ + d * H_ + j;
    hout32[o] = hn;
    hout16[o] = (_Float16)hn;
  }
}

// ---------------------------------------------------------------------------
// Fully fused token matching (the 137-GFLOP hot loop).
// Block = (side, b, i): builds the full 64x512 f16 |L_i - R_j| tile in LDS
// (64 KB dynamic; WGP has 320 KB), runs the 512x512 highway (Wn,Wg) via WMMA
// with A-frags from LDS.  Each B-fragment is amortized over 4 M-tiles + the
// n/g pair (8 WMMAs per 2 KB of weights -> 256 B/WMMA of cache traffic).
// Fuses relu/sigmoid/highway-mix, lint_W score reduction, masked softmax over
// the 64 opposite-side tokens, and the weighted sum back over the LDS tile.
// grid = 2*16*64 = 2048 blocks, 256 threads (8 waves x 4 N-chunks each).
// (lint_b is a softmax-invariant shift and is omitted.)
// ---------------------------------------------------------------------------
__global__ void k_token_match(const float* __restrict__ emb,    // (4,16,32,512)
                              const _Float16* __restrict__ Wn16,
                              const _Float16* __restrict__ Wg16,
                              const float* __restrict__ bn, const float* __restrict__ bg,
                              const float* __restrict__ lintW,
                              const int* __restrict__ maskL, const int* __restrict__ maskR,
                              float* __restrict__ cmpOut) {     // (2,16,64,512)
  extern __shared__ __align__(16) char smem[];
  _Float16* cm = (_Float16*)smem;               // 64 x 512 f16
  float* sc = (float*)(smem + 64 * 512 * 2);    // 64 scores
  float* w  = sc + 64;                          // 64 softmax weights

  int id = blockIdx.x;
  int i  = id & 63; id >>= 6;
  int b  = id & 15; id >>= 4;
  int side = id;  // 0: X=left,Y=right ; 1: swapped

  auto embAt = [&](int s, int tok, int k) -> float {
    return emb[((((long)(s * 2 + (tok >> 5))) * B_ + b) * LF_ + (tok & 31)) * H2_ + k];
  };

  for (int idx = threadIdx.x; idx < 64 * 512; idx += blockDim.x) {
    int jl = idx >> 9, k = idx & 511;
    cm[jl * 512 + k] = (_Float16)fabsf(embAt(side, i, k) - embAt(1 - side, jl, k));
  }
  if (threadIdx.x < 64) sc[threadIdx.x] = 0.0f;
  __syncthreads();

  int lane = threadIdx.x & 31, wave = threadIdx.x >> 5;
#pragma unroll 1
  for (int q = 0; q < 4; ++q) {
    int n0 = (wave + q * 8) << 4;
    v8f an[4] = {{}, {}, {}, {}};
    v8f ag[4] = {{}, {}, {}, {}};
    for (int k0 = 0; k0 < 512; k0 += 32) {
      v16h bnf = load_b_frag(Wn16, 512, n0, k0, lane);
      v16h bgf = load_b_frag(Wg16, 512, n0, k0, lane);
      if (k0 + 32 < 512) {  // prefetch next K-slice of the weight rows
        int n = n0 + (lane & 15);
        __builtin_prefetch(Wn16 + (long)n * 512 + k0 + 32, 0, 1);
        __builtin_prefetch(Wg16 + (long)n * 512 + k0 + 32, 0, 1);
      }
#pragma unroll
      for (int mt = 0; mt < 4; ++mt) {
        v16h a = load_a_frag(cm, 512, mt * 16, k0, lane);  // ds_load path
        an[mt] = wmma_f16(a, bnf, an[mt]);
        ag[mt] = wmma_f16(a, bgf, ag[mt]);
      }
    }
    int ncol = n0 + (lane & 15);
    float bnv = bn[ncol], bgv = bg[ncol], lw = lintW[ncol];
    int mb = (lane >> 4) << 3;
#pragma unroll
    for (int mt = 0; mt < 4; ++mt)
#pragma unroll
      for (int r = 0; r < 8; ++r) {
        int m = mt * 16 + mb + r;
        float nval = fmaxf(an[mt][r] + bnv, 0.0f);
        float g = sig_(ag[mt][r] + bgv);
        float hw = nval * g + (1.0f - g) * (float)cm[m * 512 + ncol];
        atomicAdd(&sc[m], hw * lw);
      }
  }
  __syncthreads();

  const int* maskY = side ? maskL : maskR;
  if (threadIdx.x == 0) {
    float mx = -1e30f;
    for (int j = 0; j < 64; ++j) {
      float v = sc[j] + (maskY[b * T_ + j] ? 0.0f : -1e10f);
      w[j] = v;
      mx = fmaxf(mx, v);
    }
    float sum = 0.0f;
    for (int j = 0; j < 64; ++j) { w[j] = expf(w[j] - mx); sum += w[j]; }
    float inv = 1.0f / sum;
    for (int j = 0; j < 64; ++j) w[j] *= inv;
  }
  __syncthreads();

  for (int k = threadIdx.x; k < H2_; k += blockDim.x) {
    float acc = 0.0f;
    for (int j = 0; j < 64; ++j) acc += w[j] * (float)cm[j * 512 + k];
    cmpOut[(((long)side * B_ + b) * T_ + i) * H2_ + k] = acc;
  }
}

// ---------------------------------------------------------------------------
// Attribute matching -> concat (B, 2048) in f32 + f16.
// Block per (side,b,field) = 64 blocks.
// ---------------------------------------------------------------------------
__global__ void k_attr(const float* __restrict__ emb, const float* __restrict__ cmpOut,
                       const int* __restrict__ maskL, const int* __restrict__ maskR,
                       const float* __restrict__ attrL, const float* __restrict__ attrR,
                       const float* __restrict__ emptyAttr,
                       float* __restrict__ concat32, _Float16* __restrict__ concat16) {
  __shared__ float w[32];
  __shared__ int amask;
  int id = blockIdx.x;
  int fld = id & 1; id >>= 1;
  int b = id & 15;  id >>= 4;
  int side = id;
  const float* attrW = (side ? attrR : attrL) + fld * H2_;
  const int* maskS = side ? maskR : maskL;
  int lane = threadIdx.x & 31, wave = threadIdx.x >> 5;
  for (int q = 0; q < 4; ++q) {
    int t = wave * 4 + q;
    const float* row = emb + ((((long)(side * 2 + fld)) * B_ + b) * LF_ + t) * H2_;
    float s = 0.0f;
    for (int k = lane; k < H2_; k += 32) s += row[k] * attrW[k];
    for (int off = 16; off > 0; off >>= 1) s += __shfl_down(s, off, 32);
    if (lane == 0) w[t] = s;
  }
  __syncthreads();
  if (threadIdx.x == 0) {
    int any = 0;
    float mx = -1e30f;
    for (int t = 0; t < 32; ++t) {
      int mk = maskS[b * T_ + fld * LF_ + t];
      any |= mk;
      float v = w[t] + (mk ? 0.0f : -1e10f);
      w[t] = v;
      mx = fmaxf(mx, v);
    }
    float sum = 0.0f;
    for (int t = 0; t < 32; ++t) { w[t] = expf(w[t] - mx); sum += w[t]; }
    float inv = 1.0f / sum;
    for (int t = 0; t < 32; ++t) w[t] *= inv;
    amask = any;
  }
  __syncthreads();
  for (int k = threadIdx.x; k < H2_; k += blockDim.x) {
    float v;
    if (amask) {
      float acc = 0.0f;
      for (int t = 0; t < 32; ++t)
        acc += w[t] * cmpOut[(((long)side * B_ + b) * T_ + fld * LF_ + t) * H2_ + k];
      v = acc;
    } else {
      v = emptyAttr[k];
    }
    long o = (long)b * DE_ + (side * 2 + fld) * H2_ + k;
    concat32[o] = v;
    concat16[o] = (_Float16)v;
  }
}

// ---------------------------------------------------------------------------
// Final 2048x2048 highway via WMMA.  16 blocks x 8 waves = 128 N-tiles; M=16.
// ---------------------------------------------------------------------------
__global__ void k_highway_final(const _Float16* __restrict__ A16,
                                const float* __restrict__ A32,
                                const _Float16* __restrict__ Wn16,
                                const _Float16* __restrict__ Wg16,
                                const float* __restrict__ bn, const float* __restrict__ bg,
                                float* __restrict__ out) {
  int lane = threadIdx.x & 31, wave = threadIdx.x >> 5;
  int n0 = (blockIdx.x * 8 + wave) << 4;
  v8f an = {}, ag = {};
  for (int k0 = 0; k0 < DE_; k0 += 32) {
    v16h a = load_a_frag(A16, DE_, 0, k0, lane);
    an = wmma_f16(a, load_b_frag(Wn16, DE_, n0, k0, lane), an);
    ag = wmma_f16(a, load_b_frag(Wg16, DE_, n0, k0, lane), ag);
  }
  int n = n0 + (lane & 15);
  float bnv = bn[n], bgv = bg[n];
  int mb = (lane >> 4) << 3;
#pragma unroll
  for (int r = 0; r < 8; ++r) {
    int m = mb + r;
    float nval = fmaxf(an[r] + bnv, 0.0f);
    float g = sig_(ag[r] + bgv);
    out[(long)m * DE_ + n] = nval * g + (1.0f - g) * A32[(long)m * DE_ + n];
  }
}

// ---------------------------------------------------------------------------
// hw @ line_W^T + line_b, then log_softmax over 2 classes.  1 block.
// ---------------------------------------------------------------------------
__global__ void k_logits(const float* __restrict__ hw, const float* __restrict__ lineW,
                         const float* __restrict__ lineB, float* __restrict__ out) {
  __shared__ float lg[16][2];
  int lane = threadIdx.x & 31, wave = threadIdx.x >> 5;
  for (int q = 0; q < 4; ++q) {
    int p = wave * 4 + q;  // 0..31 = (b,c)
    int b = p >> 1, c = p & 1;
    const float* row = hw + (long)b * DE_;
    const float* wr = lineW + (long)c * DE_;
    float s = 0.0f;
    for (int k = lane; k < DE_; k += 32) s += row[k] * wr[k];
    for (int off = 16; off > 0; off >>= 1) s += __shfl_down(s, off, 32);
    if (lane == 0) lg[b][c] = s + lineB[c];
  }
  __syncthreads();
  if (threadIdx.x < 16) {
    int b = threadIdx.x;
    float x0 = lg[b][0], x1 = lg[b][1];
    float m = fmaxf(x0, x1);
    float l = m + logf(expf(x0 - m) + expf(x1 - m));
    out[b * 2 + 0] = x0 - l;
    out[b * 2 + 1] = x1 - l;
  }
}

// ---------------------------------------------------------------------------
// Host orchestration
// ---------------------------------------------------------------------------
extern "C" void kernel_launch(void* const* d_in, const int* in_sizes, int n_in,
                              void* d_out, int out_size, void* d_ws, size_t ws_size,
                              hipStream_t stream) {
  (void)in_sizes; (void)n_in; (void)out_size; (void)ws_size;
  const float* left_f0  = (const float*)d_in[0];
  const float* left_f1  = (const float*)d_in[1];
  const float* right_f0 = (const float*)d_in[2];
  const float* right_f1 = (const float*)d_in[3];
  const float* gru_Wih0 = (const float*)d_in[4];
  const float* gru_Whh0 = (const float*)d_in[5];
  const float* gru_bih0 = (const float*)d_in[6];
  const float* gru_bhh0 = (const float*)d_in[7];
  const float* gru_Wih12 = (const float*)d_in[8];
  const float* gru_Whh12 = (const float*)d_in[9];
  const float* gru_bih12 = (const float*)d_in[10];
  const float* gru_bhh12 = (const float*)d_in[11];
  const float* hwt_Wn = (const float*)d_in[12];
  const float* hwt_bn = (const float*)d_in[13];
  const float* hwt_Wg = (const float*)d_in[14];
  const float* hwt_bg = (const float*)d_in[15];
  const float* lint_W = (const float*)d_in[16];
  const float* attr_left = (const float*)d_in[18];
  const float* attr_right = (const float*)d_in[19];
  const float* empty_attr = (const float*)d_in[20];
  const float* hwe_Wn = (const float*)d_in[21];
  const float* hwe_bn = (const float*)d_in[22];
  const float* hwe_Wg = (const float*)d_in[23];
  const float* hwe_bg = (const float*)d_in[24];
  const float* line_W = (const float*)d_in[25];
  const float* line_b = (const float*)d_in[26];

  char* ws = (char*)d_ws;
  size_t off = 0;
  auto alloc = [&](size_t bytes) -> void* {
    off = (off + 255) & ~(size_t)255;
    void* p = ws + off;
    off += bytes;
    return p;
  };
  auto allocH = [&](long n) { return (_Float16*)alloc((size_t)n * 2); };
  auto allocF = [&](long n) { return (float*)alloc((size_t)n * 4); };

  _Float16* x16      = allocH(4L * 512 * 768);
  _Float16* wih0_16  = allocH(2L * 768 * 768);
  _Float16* whh0_16  = allocH(2L * 768 * 256);
  _Float16* wih12_16 = allocH(4L * 768 * 512);
  _Float16* whh12_16 = allocH(4L * 768 * 256);
  _Float16* hwtWn16  = allocH(512L * 512);
  _Float16* hwtWg16  = allocH(512L * 512);
  _Float16* hweWn16  = allocH((long)DE_ * DE_);
  _Float16* hweWg16  = allocH((long)DE_ * DE_);
  _Float16* hA16     = allocH(4L * 512 * 512);
  _Float16* hB16     = allocH(4L * 512 * 512);
  _Float16* h16a     = allocH(8L * 16 * 256);
  _Float16* h16b     = allocH(8L * 16 * 256);
  _Float16* concat16 = allocH((long)B_ * DE_);
  float* xi       = allocF(8L * 512 * 768);
  float* hA32     = allocF(4L * 512 * 512);
  float* hB32     = allocF(4L * 512 * 512);
  float* h32a     = allocF(8L * 16 * 256);
  float* h32b     = allocF(8L * 16 * 256);
  float* cmpB     = allocF(2L * 16 * 64 * 512);
  float* concat32 = allocF((long)B_ * DE_);
  float* hwbuf    = allocF((long)B_ * DE_);
  int* maskL = (int*)alloc(2L * 16 * 64 * 4);
  int* maskR = maskL + 16 * 64;

  // Allow 64KB+ dynamic LDS for the fused token-matching kernel (WGP: 320KB).
  hipFuncSetAttribute(reinterpret_cast<const void*>(k_token_match),
                      hipFuncAttributeMaxDynamicSharedMemorySize, TM_SMEM_BYTES);

  auto conv = [&](const float* s, _Float16* d, long n) {
    long blocks = (n + 255) / 256;
    if (blocks > 4096) blocks = 4096;
    k_f2h<<<(int)blocks, 256, 0, stream>>>(s, d, n);
  };
  conv(left_f0,  x16 + 0L * 512 * 768, 512L * 768);
  conv(left_f1,  x16 + 1L * 512 * 768, 512L * 768);
  conv(right_f0, x16 + 2L * 512 * 768, 512L * 768);
  conv(right_f1, x16 + 3L * 512 * 768, 512L * 768);
  conv(gru_Wih0,  wih0_16,  2L * 768 * 768);
  conv(gru_Whh0,  whh0_16,  2L * 768 * 256);
  conv(gru_Wih12, wih12_16, 4L * 768 * 512);
  conv(gru_Whh12, whh12_16, 4L * 768 * 256);
  conv(hwt_Wn, hwtWn16, 512L * 512);
  conv(hwt_Wg, hwtWg16, 512L * 512);
  conv(hwe_Wn, hweWn16, (long)DE_ * DE_);
  conv(hwe_Wg, hweWg16, (long)DE_ * DE_);

  k_masks<<<4 * 16 * 32, 256, 0, stream>>>(left_f0, left_f1, right_f0, right_f1,
                                           maskL, maskR);

  // ---- Layer 0 (input K = 768): 8 M-groups x 48 N-tiles = 384 jobs / 8 waves ----
  k_inproj<<<dim3(48, 8), 256, 0, stream>>>(x16, 512L * 768, wih0_16, 768L * 768,
                                            gru_bih0, 768, xi, 512L * 768,
                                            512, 768, 768);
  k_zero_h<<<64, 256, 0, stream>>>(h32a, h16a, 8L * 16 * 256);
  for (int t = 0; t < LF_; ++t) {
    float* hp = (t & 1) ? h32b : h32a;
    float* hn = (t & 1) ? h32a : h32b;
    _Float16* hp16 = (t & 1) ? h16b : h16a;
    _Float16* hn16 = (t & 1) ? h16a : h16b;
    k_gru_step<<<32, 128, 0, stream>>>(xi, whh0_16, 768L * 256, gru_bhh0, 768,
                                       hp, hn, hp16, hn16, hA32, hA16, t);
  }

  // ---- Layers 1..2 (input K = 512) ----
  for (int l = 0; l < 2; ++l) {
    const _Float16* hin16 = l ? hB16 : hA16;
    float* hout32 = l ? hA32 : hB32;
    _Float16* hout16 = l ? hA16 : hB16;
    k_inproj<<<dim3(48, 8), 256, 0, stream>>>(hin16, 512L * 512,
                                              wih12_16 + (long)l * 2 * 768 * 512, 768L * 512,
                                              gru_bih12 + (long)l * 2 * 768, 768,
                                              xi, 512L * 768, 512, 768, 512);
    k_zero_h<<<64, 256, 0, stream>>>(h32a, h16a, 8L * 16 * 256);
    for (int t = 0; t < LF_; ++t) {
      float* hp = (t & 1) ? h32b : h32a;
      float* hn = (t & 1) ? h32a : h32b;
      _Float16* hp16 = (t & 1) ? h16b : h16a;
      _Float16* hn16 = (t & 1) ? h16a : h16b;
      k_gru_step<<<32, 128, 0, stream>>>(xi, whh12_16 + (long)l * 2 * 768 * 256, 768L * 256,
                                         gru_bhh12 + (long)l * 2 * 768, 768,
                                         hp, hn, hp16, hn16, hout32, hout16, t);
    }
  }
  // Final embeddings now in hA32 (4,16,32,512).

  k_token_match<<<2 * 16 * 64, 256, TM_SMEM_BYTES, stream>>>(
      hA32, hwtWn16, hwtWg16, hwt_bn, hwt_bg, lint_W, maskL, maskR, cmpB);
  k_attr<<<64, 256, 0, stream>>>(hA32, cmpB, maskL, maskR, attr_left, attr_right,
                                 empty_attr, concat32, concat16);
  k_highway_final<<<16, 256, 0, stream>>>(concat16, concat32, hweWn16, hweWg16,
                                          hwe_bn, hwe_bg, hwbuf);
  k_logits<<<1, 256, 0, stream>>>(hwbuf, line_W, line_b, (float*)d_out);
}
